// BackboneUpdate_87935160418349
// MI455X (gfx1250) — matrix operand order repaired and checked by reading, not compiled, not run on hardware
//
#include <hip/hip_runtime.h>
#include <hip/hip_bf16.h>
#include <math.h>
#include <stdint.h>

typedef __attribute__((ext_vector_type(16))) _Float16 v16h;
typedef __attribute__((ext_vector_type(8)))  _Float16 v8h;
typedef __attribute__((ext_vector_type(8)))  float    v8f;
typedef int v4i __attribute__((vector_size(16)));

#define NN    4096
#define LTOT  9
#define BBCN  32
#define SPHN  35
#define KNNE  20
#define LRKE  40
#define DEGE  60
#define EE    (NN*DEGE)        // 245760 edges
#define HEADSN 8
#define VCN    8
#define HVN    64              // HEADS*VC
#define AINP   128             // a_in padded (102 -> 128)
#define HCN    32
#define BIGF   1.0e30f

// Optional CDNA5 async global->LDS DMA (ASYNCcnt path), guarded by __has_builtin
#if defined(__has_builtin)
#if __has_builtin(__builtin_amdgcn_global_load_async_to_lds_b128) && \
    __has_builtin(__builtin_amdgcn_s_wait_asynccnt)
#define HAVE_ASYNC_LDS 1
#endif
#endif

#ifdef HAVE_ASYNC_LDS
typedef __attribute__((address_space(1))) v4i gas_v4i;
typedef __attribute__((address_space(3))) v4i las_v4i;
__device__ inline void async_copy16(const void* g, void* l) {
  __builtin_amdgcn_global_load_async_to_lds_b128((gas_v4i*)g, (las_v4i*)l, 0, 0);
}
#endif

// ---------------------------------------------------------------------------
// WMMA fragment helpers (contiguous k = 16*(lane/16) + element for A and B, wave32)
// ---------------------------------------------------------------------------
// A: 16xK row-major tile in LDS; per lane the 16 halves live in two contiguous
// 16B groups -> two b128 LDS loads.
__device__ inline v16h frag_a_rowmajor(const _Float16* A, int lda, int lane) {
  int m    = lane & 15;
  int koff = (lane >> 4) * 16;
  const _Float16* base = A + m * lda + koff;
  v8h lo = *(const v8h*)(base);
  v8h hi = *(const v8h*)(base + 8);
  return __builtin_shufflevector(lo, hi, 0,1,2,3,4,5,6,7,8,9,10,11,12,13,14,15);
}

// B: stored swizzled in LDS so each lane reads its 16 halves contiguously
// (32B per lane -> two b128 LDS loads). Element h of the fragment corresponds
// to row k = 16*(lane>=16) + h (within the 32-row K chunk), col n = lane&15.
__device__ inline v16h frag_b_swz(const _Float16* Bs, int lane) {
  const v8h* p = (const v8h*)(Bs + lane * 16);
  v8h lo = p[0];
  v8h hi = p[1];
  return __builtin_shufflevector(lo, hi, 0,1,2,3,4,5,6,7,8,9,10,11,12,13,14,15);
}

__device__ inline int so3_deg(int l) { return (l == 0) ? 0 : ((l < 4) ? 1 : 2); }

__device__ inline float siluf(float x) { return x / (1.0f + expf(-x)); }
__device__ inline float softplusf(float x) { return (x > 20.0f) ? x : log1pf(expf(x)); }

__device__ inline float gumbel_hash(unsigned h) {
  h ^= h >> 16; h *= 0x7feb352dU; h ^= h >> 15; h *= 0x846ca68bU; h ^= h >> 16;
  float u = (float)(h >> 8) * (1.0f / 16777216.0f) + 1.0e-7f;
  return -logf(-logf(u));
}

// ---------------------------------------------------------------------------
// 1) squared norms
// ---------------------------------------------------------------------------
__global__ void k_x2(const float* __restrict__ X, float* __restrict__ x2) {
  int i = blockIdx.x * blockDim.x + threadIdx.x;
  if (i < NN) {
    float a = X[i*3+0], b = X[i*3+1], c = X[i*3+2];
    x2[i] = a*a + b*b + c*c;
  }
}

// ---------------------------------------------------------------------------
// 2) per-row distance + KNN/long-range selection (row resident in LDS)
// ---------------------------------------------------------------------------
__global__ void k_topk(const float* __restrict__ X, const float* __restrict__ x2,
                       const int* __restrict__ batch, const unsigned char* __restrict__ x_mask,
                       int* __restrict__ src) {
  __shared__ float dbuf[NN];
  __shared__ float rv[256];
  __shared__ int   ri[256];
  const int i = blockIdx.x;
  const int tid = threadIdx.x;
  const int bi = batch[i];
  const float xi0 = X[i*3+0], xi1 = X[i*3+1], xi2 = X[i*3+2];
  const float x2i = x2[i];
  const int mi = x_mask[i];

  for (int j = tid; j < NN; j += 256) {
    float d2 = x2i + x2[j] - 2.0f*(xi0*X[j*3+0] + xi1*X[j*3+1] + xi2*X[j*3+2]);
    d2 = fmaxf(d2, 0.0f);
    bool invalid = (batch[j] != bi) || (j == i) || mi || x_mask[j];
    dbuf[j] = invalid ? BIGF : d2;
  }
  __syncthreads();

  // KNN: 20 argmin passes
  for (int t = 0; t < KNNE; ++t) {
    float bv = 3.0e38f; int bj = 0;
    for (int j = tid; j < NN; j += 256) {
      float v = dbuf[j];
      if (v < bv) { bv = v; bj = j; }
    }
    rv[tid] = bv; ri[tid] = bj;
    __syncthreads();
    for (int s = 128; s > 0; s >>= 1) {
      if (tid < s && rv[tid + s] < rv[tid]) { rv[tid] = rv[tid + s]; ri[tid] = ri[tid + s]; }
      __syncthreads();
    }
    if (tid == 0) {
      int w = ri[0];
      src[i*KNNE + t] = w;
      dbuf[w] = BIGF;
    }
    __syncthreads();
  }

  // logits + gumbel (hash-based) in place
  for (int j = tid; j < NN; j += 256) {
    float v = dbuf[j];
    float lg = (v >= 0.5f * BIGF) ? -BIGF
                                  : (-1.5f * logf(v + 1.0e-6f) + gumbel_hash((unsigned)(i*NN + j)));
    dbuf[j] = lg;
  }
  __syncthreads();

  // long-range: 40 argmax passes
  for (int t = 0; t < LRKE; ++t) {
    float bv = -3.0e38f; int bj = 0;
    for (int j = tid; j < NN; j += 256) {
      float v = dbuf[j];
      if (v > bv) { bv = v; bj = j; }
    }
    rv[tid] = bv; ri[tid] = bj;
    __syncthreads();
    for (int s = 128; s > 0; s >>= 1) {
      if (tid < s && rv[tid + s] > rv[tid]) { rv[tid] = rv[tid + s]; ri[tid] = ri[tid + s]; }
      __syncthreads();
    }
    if (tid == 0) {
      int w = ri[0];
      src[NN*KNNE + i*LRKE + t] = w;
      dbuf[w] = -3.0e38f;
    }
    __syncthreads();
  }
}

// ---------------------------------------------------------------------------
// 3) build node features: inv (f32, N x 35) and node_h (f16, N x 9 x 64 padded)
// ---------------------------------------------------------------------------
__global__ void k_node(const float* __restrict__ bb_features, const float* __restrict__ bb_rel,
                       const unsigned char* __restrict__ noising,
                       float* __restrict__ inv, _Float16* __restrict__ node_h) {
  int id = blockIdx.x * blockDim.x + threadIdx.x;
  if (id >= NN * LTOT * 64) return;
  int n = id / 576;
  int r = id % 576;
  int l = r >> 6;
  int k = r & 63;
  float val = 0.0f;
  if (k < BBCN) {
    val = bb_features[n*288 + l*32 + k];
  } else if (k < SPHN) {
    if (l >= 1 && l <= 3) {
      int b = k - BBCN, d = l - 1;
      val = bb_rel[(n*3 + b)*3 + d];
    } else if (l == 0 && k == SPHN - 1) {
      val = (float)noising[n];
    }
  }
  node_h[id] = (_Float16)val;
  if (l == 0 && k < SPHN) inv[n*SPHN + k] = val;
}

// ---------------------------------------------------------------------------
// 4) edge MLP via WMMA: a_in(16x128 f16, LDS) @ W1(128x32) -> silu -> @ W2(32x8)
//    Weights staged via async global->LDS DMA when available.
// ---------------------------------------------------------------------------
__global__ void __launch_bounds__(128)
k_edge_mlp(const int* __restrict__ src, const float* __restrict__ inv,
           const float* __restrict__ X,
           const float* __restrict__ w1, const float* __restrict__ b1,
           const float* __restrict__ w2, const float* __restrict__ b2,
           float* __restrict__ logit) {
  // swizzled f16 weights: [kchunk][coltile][lane][16]
  __shared__ __attribute__((aligned(32))) _Float16 w1s[4 * 2 * 32 * 16];  // 8KB
  __shared__ __attribute__((aligned(32))) _Float16 w2s[1 * 1 * 32 * 16];  // 1KB
  __shared__ __attribute__((aligned(32))) _Float16 as[4][16][AINP];       // 16KB
  __shared__ __attribute__((aligned(32))) _Float16 hs[4][16][HCN];        // 4KB
#ifdef HAVE_ASYNC_LDS
  __shared__ __attribute__((aligned(16))) float w1raw[102 * HCN];         // 13056B
  __shared__ __attribute__((aligned(16))) float w2raw[HCN * HEADSN];      // 1KB
#endif

  const int tid  = threadIdx.x;
  const int wid  = tid >> 5;
  const int lane = tid & 31;

#ifdef HAVE_ASYNC_LDS
  // DMA raw f32 weights into LDS (ASYNCcnt-tracked), then convert+swizzle.
  {
    const int n1 = (102 * HCN * 4) / 16;   // 816 x 16B
    for (int c = tid; c < n1; c += 128)
      async_copy16((const char*)w1 + c * 16, (char*)w1raw + c * 16);
    const int n2 = (HCN * HEADSN * 4) / 16; // 64 x 16B
    for (int c = tid; c < n2; c += 128)
      async_copy16((const char*)w2 + c * 16, (char*)w2raw + c * 16);
    __builtin_amdgcn_s_wait_asynccnt(0);
  }
  __syncthreads();
  const float* w1p = w1raw;
  const float* w2p = w2raw;
#else
  const float* w1p = w1;
  const float* w2p = w2;
#endif

  // build swizzled f16 fragments of W1 (4 K-chunks x 2 col-tiles) and W2
  for (int idx = tid; idx < 4 * 2 * 32 * 16; idx += 128) {
    int h  = idx & 15;
    int ln = (idx >> 4) & 31;
    int ct = (idx >> 9) & 1;
    int kc = idx >> 10;
    int k  = kc * 32 + ((ln >> 4) << 4) + h;
    int c  = ct * 16 + (ln & 15);
    w1s[idx] = (_Float16)((k < 102) ? w1p[k * HCN + c] : 0.0f);
  }
  for (int idx = tid; idx < 32 * 16; idx += 128) {
    int h  = idx & 15;
    int ln = (idx >> 4) & 31;
    int k  = ((ln >> 4) << 4) + h;
    int c  = ln & 15;
    w2s[idx] = (_Float16)((c < HEADSN) ? w2p[k * HEADSN + c] : 0.0f);
  }

  const int tile = blockIdx.x * 4 + wid;
  const int e0   = tile * 16;

  // build a_in tile: lanes 0..15 each own one edge row
  if (lane < 16) {
    int e = e0 + lane;
    int s = src[e];
    int d = (e < NN*KNNE) ? (e / KNNE) : ((e - NN*KNNE) / LRKE);
    _Float16* arow = &as[wid][lane][0];
    for (int k = 0; k < SPHN; ++k) arow[k]        = (_Float16)inv[s*SPHN + k];
    for (int k = 0; k < SPHN; ++k) arow[SPHN + k] = (_Float16)inv[d*SPHN + k];
    float dx = X[s*3+0] - X[d*3+0];
    float dy = X[s*3+1] - X[d*3+1];
    float dz = X[s*3+2] - X[d*3+2];
    float dist = sqrtf(dx*dx + dy*dy + dz*dz);
    const float inv_sigma = 1.0f / 1.25f;
    for (int t = 0; t < 16; ++t) {
      float mu = (20.0f / 15.0f) * (float)t;
      float z = (dist - mu) * inv_sigma;
      arow[70 + t] = (_Float16)expf(-z*z);
    }
    float dd = (float)(s - d);
    const float c0 = -logf(10000.0f) / 16.0f;
    for (int t = 0; t < 8; ++t) {
      float freq = expf((float)(2*t) * c0);
      float ang  = dd * freq;
      arow[86 + t] = (_Float16)cosf(ang);
      arow[94 + t] = (_Float16)sinf(ang);
    }
    for (int k = 102; k < AINP; ++k) arow[k] = (_Float16)0.0f;
  }
  __syncthreads();

  // h = silu(A @ W1 + b1): two 16-col tiles, 4 K-chunks of 32
  for (int ct = 0; ct < 2; ++ct) {
    v8f acc = {};
#pragma unroll
    for (int kc = 0; kc < 4; ++kc) {
      v16h a = frag_a_rowmajor(&as[wid][0][0] + kc*32, AINP, lane);
      v16h b = frag_b_swz(w1s + (kc*2 + ct)*512, lane);
      acc = __builtin_amdgcn_wmma_f32_16x16x32_f16(false, a, false, b, (short)0, acc,
                                                   false, false);
    }
    int col = ct*16 + (lane & 15);
    int mb  = (lane >> 4) * 8;
    float bc = b1[col];
#pragma unroll
    for (int r = 0; r < 8; ++r) {
      hs[wid][mb + r][col] = (_Float16)siluf(acc[r] + bc);
    }
  }
  __syncthreads();

  // logit = H @ W2 + b2 (cols 0..7 valid)
  v8f acc2 = {};
  {
    v16h a = frag_a_rowmajor(&hs[wid][0][0], HCN, lane);
    v16h b = frag_b_swz(w2s, lane);
    acc2 = __builtin_amdgcn_wmma_f32_16x16x32_f16(false, a, false, b, (short)0, acc2,
                                                  false, false);
  }
  int ncol = lane & 15;
  int mb2  = (lane >> 4) * 8;
  if (ncol < HEADSN) {
    float bc = b2[ncol];
#pragma unroll
    for (int r = 0; r < 8; ++r) {
      int e = e0 + mb2 + r;
      logit[(size_t)e * HEADSN + ncol] = acc2[r] + bc;
    }
  }
}

// ---------------------------------------------------------------------------
// 5) per-(dst,head) softmax over fixed fan-in of 60 edges (alpha in place)
// ---------------------------------------------------------------------------
__global__ void k_softmax(float* __restrict__ logit) {
  int id = blockIdx.x * blockDim.x + threadIdx.x;
  if (id >= NN * HEADSN) return;
  int n = id >> 3, h = id & 7;
  float m = -3.0e38f;
  for (int j = 0; j < DEGE; ++j) {
    int e = (j < KNNE) ? (n*KNNE + j) : (NN*KNNE + n*LRKE + (j - KNNE));
    m = fmaxf(m, logit[(size_t)e*HEADSN + h]);
  }
  float Z = 0.0f;
  for (int j = 0; j < DEGE; ++j) {
    int e = (j < KNNE) ? (n*KNNE + j) : (NN*KNNE + n*LRKE + (j - KNNE));
    Z += expf(logit[(size_t)e*HEADSN + h] - m);
  }
  float invZ = 1.0f / (Z + 1.0e-9f);
  for (int j = 0; j < DEGE; ++j) {
    int e = (j < KNNE) ? (n*KNNE + j) : (NN*KNNE + n*LRKE + (j - KNNE));
    size_t ix = (size_t)e*HEADSN + h;
    logit[ix] = expf(logit[ix] - m) * invZ;
  }
}

// ---------------------------------------------------------------------------
// 6) v = so3_linear(node, wv, bv) via WMMA  (Cin 35->64 pad, Cout 64)
// ---------------------------------------------------------------------------
__global__ void __launch_bounds__(32)
k_so3_wv(const _Float16* __restrict__ node_h, const float* __restrict__ wv,
         const float* __restrict__ bv, float* __restrict__ vout) {
  __shared__ __attribute__((aligned(32))) _Float16 bw[2 * 32 * 16];  // this block's col-tile
  __shared__ __attribute__((aligned(32))) _Float16 at[16 * 64];
  int lane = threadIdx.x;
  int n0 = blockIdx.x * 16;
  int l  = blockIdx.y;
  int ct = blockIdx.z;
  int ld = so3_deg(l);
  for (int idx = lane; idx < 2 * 32 * 16; idx += 32) {
    int h  = idx & 15;
    int ln = (idx >> 4) & 31;
    int kc = idx >> 9;
    int k  = kc * 32 + ((ln >> 4) << 4) + h;
    int c  = ct * 16 + (ln & 15);
    bw[idx] = (_Float16)((k < SPHN) ? wv[ld*SPHN*HVN + k*HVN + c] : 0.0f);
  }
  for (int idx = lane; idx < 16*64; idx += 32) {
    int m = idx >> 6, k = idx & 63;
    at[idx] = node_h[(size_t)(n0+m)*576 + l*64 + k];
  }
  __syncthreads();
  v8f acc = {};
#pragma unroll
  for (int kc = 0; kc < 2; ++kc) {
    v16h a = frag_a_rowmajor(at + kc*32, 64, lane);
    v16h b = frag_b_swz(bw + kc*512, lane);
    acc = __builtin_amdgcn_wmma_f32_16x16x32_f16(false, a, false, b, (short)0, acc,
                                                 false, false);
  }
  int c  = ct*16 + (lane & 15);
  int mb = (lane >> 4) * 8;
#pragma unroll
  for (int r = 0; r < 8; ++r) {
    float val = acc[r] + ((l == 0) ? bv[c] : 0.0f);
    vout[(size_t)(n0+mb+r)*576 + l*64 + c] = val;
  }
}

// ---------------------------------------------------------------------------
// 7) gather + alpha-weight + aggregate over 60 edges (L2-resident v table)
// ---------------------------------------------------------------------------
__global__ void k_agg(const int* __restrict__ src, const float* __restrict__ alpha,
                      const float* __restrict__ vbuf,
                      float* __restrict__ agg, _Float16* __restrict__ agg_h) {
  int id = blockIdx.x * blockDim.x + threadIdx.x;
  if (id >= NN * 576) return;
  int n = id / 576;
  int r = id % 576;
  int l = r >> 6, c = r & 63;
  int h = c >> 3;
  float s = 0.0f;
  for (int j = 0; j < DEGE; ++j) {
    int e = (j < KNNE) ? (n*KNNE + j) : (NN*KNNE + n*LRKE + (j - KNNE));
    int si = src[e];
    const float* vp = &vbuf[(size_t)si*576 + l*64 + c];
    __builtin_prefetch(vp, 0, 0);
    s += alpha[(size_t)e*HEADSN + h] * (*vp);
  }
  agg[id]   = s;
  agg_h[id] = (_Float16)s;
}

// ---------------------------------------------------------------------------
// 8) out = so3_linear(agg, wo, bo) via WMMA (Cin 64, Cout 32); also f16 copy
// ---------------------------------------------------------------------------
__global__ void __launch_bounds__(32)
k_so3_wo(const _Float16* __restrict__ agg_h, const float* __restrict__ wo,
         const float* __restrict__ bo, float* __restrict__ outf,
         _Float16* __restrict__ out_h) {
  __shared__ __attribute__((aligned(32))) _Float16 bw[2 * 32 * 16];
  __shared__ __attribute__((aligned(32))) _Float16 at[16 * 64];
  int lane = threadIdx.x;
  int n0 = blockIdx.x * 16;
  int l  = blockIdx.y;
  int ct = blockIdx.z;
  int ld = so3_deg(l);
  for (int idx = lane; idx < 2 * 32 * 16; idx += 32) {
    int h  = idx & 15;
    int ln = (idx >> 4) & 31;
    int kc = idx >> 9;
    int k  = kc * 32 + ((ln >> 4) << 4) + h;
    int c  = ct * 16 + (ln & 15);
    bw[idx] = (_Float16)wo[ld*64*32 + k*32 + c];
  }
  for (int idx = lane; idx < 16*64; idx += 32) {
    int m = idx >> 6, k = idx & 63;
    at[idx] = agg_h[(size_t)(n0+m)*576 + l*64 + k];
  }
  __syncthreads();
  v8f acc = {};
#pragma unroll
  for (int kc = 0; kc < 2; ++kc) {
    v16h a = frag_a_rowmajor(at + kc*32, 64, lane);
    v16h b = frag_b_swz(bw + kc*512, lane);
    acc = __builtin_amdgcn_wmma_f32_16x16x32_f16(false, a, false, b, (short)0, acc,
                                                 false, false);
  }
  int c  = ct*16 + (lane & 15);
  int mb = (lane >> 4) * 8;
#pragma unroll
  for (int r = 0; r < 8; ++r) {
    float val = acc[r] + ((l == 0) ? bo[c] : 0.0f);
    size_t ix = (size_t)(n0+mb+r)*288 + l*32 + c;
    outf[ix]  = val;
    out_h[ix] = (_Float16)val;
  }
}

// ---------------------------------------------------------------------------
// 9) gate = silu(out0 @ wg + bg)
// ---------------------------------------------------------------------------
__global__ void k_gate(const float* __restrict__ outf, const float* __restrict__ wg,
                       const float* __restrict__ bg, float* __restrict__ gate) {
  int id = blockIdx.x * blockDim.x + threadIdx.x;
  if (id >= NN * 32) return;
  int n = id >> 5, c = id & 31;
  float s = bg[c];
  for (int k = 0; k < 32; ++k) s += outf[(size_t)n*288 + k] * wg[k*32 + c];
  gate[id] = siluf(s);
}

// ---------------------------------------------------------------------------
// 10) out += so3_linear(out, wf, bf) * gate   via WMMA (Cin 32, Cout 32)
// ---------------------------------------------------------------------------
__global__ void __launch_bounds__(32)
k_so3_wf(const _Float16* __restrict__ out_h, const float* __restrict__ wf,
         const float* __restrict__ bf, const float* __restrict__ gate,
         float* __restrict__ outf) {
  __shared__ __attribute__((aligned(32))) _Float16 bw[32 * 16];
  __shared__ __attribute__((aligned(32))) _Float16 at[16 * 32];
  int lane = threadIdx.x;
  int n0 = blockIdx.x * 16;
  int l  = blockIdx.y;
  int ct = blockIdx.z;
  int ld = so3_deg(l);
  for (int idx = lane; idx < 32 * 16; idx += 32) {
    int h  = idx & 15;
    int ln = (idx >> 4) & 31;
    int k  = ((ln >> 4) << 4) + h;
    int c  = ct * 16 + (ln & 15);
    bw[idx] = (_Float16)wf[ld*32*32 + k*32 + c];
  }
  for (int idx = lane; idx < 16*32; idx += 32) {
    int m = idx >> 5, k = idx & 31;
    at[idx] = out_h[(size_t)(n0+m)*288 + l*32 + k];
  }
  __syncthreads();
  v8f acc = {};
  {
    v16h a = frag_a_rowmajor(at, 32, lane);
    v16h b = frag_b_swz(bw, lane);
    acc = __builtin_amdgcn_wmma_f32_16x16x32_f16(false, a, false, b, (short)0, acc,
                                                 false, false);
  }
  int c  = ct*16 + (lane & 15);
  int mb = (lane >> 4) * 8;
#pragma unroll
  for (int r = 0; r < 8; ++r) {
    float tmp = acc[r] + ((l == 0) ? bf[c] : 0.0f);
    size_t ix = (size_t)(n0+mb+r)*288 + l*32 + c;
    outf[ix] = outf[ix] + tmp * gate[(size_t)(n0+mb+r)*32 + c];
  }
}

// ---------------------------------------------------------------------------
// 11) final: coordinate + backbone updates (degree-1 rows only)
// ---------------------------------------------------------------------------
__global__ void k_final(const float* __restrict__ outf, const float* __restrict__ X,
                        const float* __restrict__ bb_rel, const unsigned char* __restrict__ noising,
                        const float* __restrict__ w_xca, const float* __restrict__ w_gate,
                        const float* __restrict__ b_gate, const float* __restrict__ w_bb,
                        float* __restrict__ newX, float* __restrict__ newbb) {
  int n = blockIdx.x * blockDim.x + threadIdx.x;
  if (n >= NN) return;
  const float* on = &outf[(size_t)n*288];
  float gz = b_gate[0];
  for (int c = 0; c < 32; ++c) gz += on[c] * w_gate[c];
  float g2 = softplusf(gz);
  int msk = noising[n];
  for (int d = 0; d < 3; ++d) {
    float v = 0.0f;
    for (int c = 0; c < 32; ++c) v += on[(1+d)*32 + c] * w_xca[1*32 + c];
    newX[n*3 + d] = X[n*3 + d] + (msk ? v * g2 : 0.0f);
  }
  for (int b = 0; b < 3; ++b) {
    for (int d = 0; d < 3; ++d) {
      float v = 0.0f;
      for (int c = 0; c < 32; ++c) v += on[(1+d)*32 + c] * w_bb[(1*32 + c)*3 + b];
      newbb[n*9 + b*3 + d] = bb_rel[n*9 + b*3 + d] + (msk ? v : 0.0f);
    }
  }
}

// ---------------------------------------------------------------------------
// launcher
// ---------------------------------------------------------------------------
extern "C" void kernel_launch(void* const* d_in, const int* in_sizes, int n_in,
                              void* d_out, int out_size, void* d_ws, size_t ws_size,
                              hipStream_t stream) {
  (void)in_sizes; (void)n_in; (void)out_size; (void)ws_size;
  const float*         X_ca     = (const float*)d_in[0];
  const float*         bb_rel   = (const float*)d_in[1];
  const float*         bb_feat  = (const float*)d_in[2];
  const int*           batch    = (const int*)d_in[3];
  const unsigned char* x_mask   = (const unsigned char*)d_in[4];
  const unsigned char* noising  = (const unsigned char*)d_in[5];
  const float*         attn_w1  = (const float*)d_in[6];
  const float*         attn_b1  = (const float*)d_in[7];
  const float*         attn_w2  = (const float*)d_in[8];
  const float*         attn_b2  = (const float*)d_in[9];
  const float*         wv       = (const float*)d_in[10];
  const float*         bv       = (const float*)d_in[11];
  const float*         wo       = (const float*)d_in[12];
  const float*         bo       = (const float*)d_in[13];
  const float*         wg       = (const float*)d_in[14];
  const float*         bg       = (const float*)d_in[15];
  const float*         wf       = (const float*)d_in[16];
  const float*         bf       = (const float*)d_in[17];
  const float*         w_xca    = (const float*)d_in[18];
  const float*         w_gate   = (const float*)d_in[20];
  const float*         b_gate   = (const float*)d_in[21];
  const float*         w_bb     = (const float*)d_in[22];

  float* out_newX  = (float*)d_out;           // N*3
  float* out_newbb = out_newX + NN*3;         // N*9
  float* out_out   = out_newbb + NN*9;        // N*9*32

  uintptr_t p = (uintptr_t)d_ws;
  auto wsalloc = [&](size_t bytes) -> void* {
    void* r = (void*)p;
    p += (bytes + 255) & ~(size_t)255;
    return r;
  };
  float*     x2     = (float*)wsalloc((size_t)NN * 4);
  int*       srcb   = (int*)wsalloc((size_t)EE * 4);
  float*     logit  = (float*)wsalloc((size_t)EE * HEADSN * 4);   // becomes alpha
  float*     inv    = (float*)wsalloc((size_t)NN * SPHN * 4);
  _Float16*  node_h = (_Float16*)wsalloc((size_t)NN * 576 * 2);
  float*     vbuf   = (float*)wsalloc((size_t)NN * 576 * 4);
  float*     aggb   = (float*)wsalloc((size_t)NN * 576 * 4);
  _Float16*  agg_h  = (_Float16*)wsalloc((size_t)NN * 576 * 2);
  _Float16*  out_h  = (_Float16*)wsalloc((size_t)NN * 288 * 2);
  float*     gate   = (float*)wsalloc((size_t)NN * 32 * 4);

  k_x2<<<(NN + 255)/256, 256, 0, stream>>>(X_ca, x2);
  k_topk<<<NN, 256, 0, stream>>>(X_ca, x2, batch, x_mask, srcb);
  k_node<<<(NN*576 + 255)/256, 256, 0, stream>>>(bb_feat, bb_rel, noising, inv, node_h);
  k_edge_mlp<<<(EE/16)/4, 128, 0, stream>>>(srcb, inv, X_ca, attn_w1, attn_b1,
                                            attn_w2, attn_b2, logit);
  k_softmax<<<(NN*HEADSN + 255)/256, 256, 0, stream>>>(logit);
  {
    dim3 g(NN/16, LTOT, 4);
    k_so3_wv<<<g, 32, 0, stream>>>(node_h, wv, bv, vbuf);
  }
  k_agg<<<(NN*576 + 255)/256, 256, 0, stream>>>(srcb, logit, vbuf, aggb, agg_h);
  {
    dim3 g(NN/16, LTOT, 2);
    k_so3_wo<<<g, 32, 0, stream>>>(agg_h, wo, bo, out_out, out_h);
  }
  k_gate<<<(NN*32 + 255)/256, 256, 0, stream>>>(out_out, wg, bg, gate);
  {
    dim3 g(NN/16, LTOT, 2);
    k_so3_wf<<<g, 32, 0, stream>>>(out_h, wf, bf, gate, out_out);
  }
  k_final<<<(NN + 255)/256, 256, 0, stream>>>(out_out, X_ca, bb_rel, noising,
                                              w_xca, w_gate, b_gate, w_bb,
                                              out_newX, out_newbb);
}
